// MultiHeadAttention_10462540333772
// MI455X (gfx1250) — compile-verified, matrix operand order checked by reading, not run on hardware
//
#include <hip/hip_runtime.h>

// ---------------------------------------------------------------------------
// MHA with head-axis softmax, fused for MI455X (gfx1250, wave32, WMMA bf16).
// Pipeline: 3x proj GEMM (f32 -> bf16; V written head-transposed) ->
//           fused attention (one barrier/iter, contiguous fragments,
//           global_prefetch of the next k-tile) -> output GEMM.
// All matrix math on v_wmma_f32_16x16x32_bf16 (f32 accumulate).
// ---------------------------------------------------------------------------

typedef __bf16 bf16_t;
typedef __attribute__((ext_vector_type(16))) __bf16 v16bf;
typedef __attribute__((ext_vector_type(8)))  __bf16 v8bf;
typedef __attribute__((ext_vector_type(8)))  float  v8f;

union V16BF { v16bf v; v8bf h[2]; };

constexpr int D_MODEL = 1024;
constexpr int S_LEN   = 2048;
constexpr int N_BATCH = 4;
constexpr int N_HEAD  = 8;
constexpr int D_HEAD  = 128;
constexpr int BS      = N_BATCH * S_LEN;   // 8192 rows for the projections

// --- WMMA 16x16x32 bf16 fragment loaders -----------------------------------
// A (16x32, MxK) 16-bit layout: lanes 0-15 hold row M=lane with K = 0..7 /
// 16..23; lanes 16-31 hold row M=lane-16 with K = 8..15 / 24..31.
// B (32x16, KxN) mirrors this with lane -> column N; for y = x*W^T, column n
// of B is row n of W, so A and B use the same row-major load pattern.

__device__ __forceinline__ v16bf frag_bf16(const bf16_t* __restrict__ base,
                                           int ld, int row0, int k0) {
  const int lane = threadIdx.x & 31;
  const int r    = lane & 15;
  const int kb   = k0 + ((lane >> 4) << 3);     // +0 lanes 0-15, +8 lanes 16-31
  const bf16_t* p = base + (size_t)(row0 + r) * ld + kb;
  V16BF u;
  u.h[0] = *(const v8bf*)(p);        // K = kb .. kb+7
  u.h[1] = *(const v8bf*)(p + 16);   // K = kb+16 .. kb+23
  return u.v;
}

__device__ __forceinline__ v16bf frag_f32(const float* __restrict__ base,
                                          int ld, int row0, int k0) {
  const int lane = threadIdx.x & 31;
  const int r    = lane & 15;
  const int kb   = k0 + ((lane >> 4) << 3);
  const float* p = base + (size_t)(row0 + r) * ld + kb;
  float4 a = *(const float4*)(p);
  float4 b = *(const float4*)(p + 4);
  float4 c = *(const float4*)(p + 16);
  float4 d = *(const float4*)(p + 20);
  v16bf o;
  o[0]  = (__bf16)a.x; o[1]  = (__bf16)a.y; o[2]  = (__bf16)a.z; o[3]  = (__bf16)a.w;
  o[4]  = (__bf16)b.x; o[5]  = (__bf16)b.y; o[6]  = (__bf16)b.z; o[7]  = (__bf16)b.w;
  o[8]  = (__bf16)c.x; o[9]  = (__bf16)c.y; o[10] = (__bf16)c.z; o[11] = (__bf16)c.w;
  o[12] = (__bf16)d.x; o[13] = (__bf16)d.y; o[14] = (__bf16)d.z; o[15] = (__bf16)d.w;
  return o;
}

// --- GEMM: Y[M,N] = X[M,K] * W[N,K]^T, 8 waves/block, wave tile 16x64 --------

__global__ __launch_bounds__(256) void gemm_nt_f32_to_bf16(
    const float* __restrict__ X, const float* __restrict__ W,
    bf16_t* __restrict__ Y) {
  const int wave = threadIdx.x >> 5;
  const int lane = threadIdx.x & 31;
  const int m0 = blockIdx.y * 128 + wave * 16;
  const int n0 = blockIdx.x * 64;
  v8f acc[4] = {};
  for (int k0 = 0; k0 < D_MODEL; k0 += 32) {
    v16bf a = frag_f32(X, D_MODEL, m0, k0);
    for (int t = 0; t < 4; ++t) {
      v16bf bf = frag_f32(W, D_MODEL, n0 + t * 16, k0);
      acc[t] = __builtin_amdgcn_wmma_f32_16x16x32_bf16(
          false, a, false, bf, (short)0, acc[t], false, false);
    }
  }
  // C/D layout: VGPR r, lane l -> (M = r + 8*(l>=16), N = l%16)
  for (int t = 0; t < 4; ++t)
    for (int r = 0; r < 8; ++r) {
      const int m = m0 + r + ((lane >> 4) << 3);
      const int n = n0 + t * 16 + (lane & 15);
      Y[(size_t)m * D_MODEL + n] = (bf16_t)acc[t][r];
    }
}

// Same GEMM, but scatters the result into head-transposed layout
// YT[((b*H + h)*DH + d) * S + s]  (m = b*S+s, n = h*DH+d), so that the
// attention kernel can read V B-fragments with contiguous 16-byte loads.
__global__ __launch_bounds__(256) void gemm_nt_f32_to_bf16_vT(
    const float* __restrict__ X, const float* __restrict__ W,
    bf16_t* __restrict__ YT) {
  const int wave = threadIdx.x >> 5;
  const int lane = threadIdx.x & 31;
  const int m0 = blockIdx.y * 128 + wave * 16;
  const int n0 = blockIdx.x * 64;
  v8f acc[4] = {};
  for (int k0 = 0; k0 < D_MODEL; k0 += 32) {
    v16bf a = frag_f32(X, D_MODEL, m0, k0);
    for (int t = 0; t < 4; ++t) {
      v16bf bf = frag_f32(W, D_MODEL, n0 + t * 16, k0);
      acc[t] = __builtin_amdgcn_wmma_f32_16x16x32_bf16(
          false, a, false, bf, (short)0, acc[t], false, false);
    }
  }
  for (int t = 0; t < 4; ++t)
    for (int r = 0; r < 8; ++r) {
      const int m  = m0 + r + ((lane >> 4) << 3);
      const int n  = n0 + t * 16 + (lane & 15);
      const int bb = m >> 11, ss = m & (S_LEN - 1);
      const int hh = n >> 7,  dd = n & (D_HEAD - 1);
      YT[(size_t)((bb * N_HEAD + hh) * D_HEAD + dd) * S_LEN + ss] =
          (bf16_t)acc[t][r];
    }
}

__global__ __launch_bounds__(256) void gemm_nt_bf16_to_f32(
    const bf16_t* __restrict__ X, const float* __restrict__ W,
    float* __restrict__ Y) {
  const int wave = threadIdx.x >> 5;
  const int lane = threadIdx.x & 31;
  const int m0 = blockIdx.y * 128 + wave * 16;
  const int n0 = blockIdx.x * 64;
  v8f acc[4] = {};
  for (int k0 = 0; k0 < D_MODEL; k0 += 32) {
    v16bf a = frag_bf16(X, D_MODEL, m0, k0);
    for (int t = 0; t < 4; ++t) {
      v16bf bf = frag_f32(W, D_MODEL, n0 + t * 16, k0);
      acc[t] = __builtin_amdgcn_wmma_f32_16x16x32_bf16(
          false, a, false, bf, (short)0, acc[t], false, false);
    }
  }
  for (int t = 0; t < 4; ++t)
    for (int r = 0; r < 8; ++r) {
      const int m = m0 + r + ((lane >> 4) << 3);
      const int n = n0 + t * 16 + (lane & 15);
      Y[(size_t)m * D_MODEL + n] = acc[t][r];
    }
}

// --- Fused attention with HEAD-axis softmax ---------------------------------
// Block = 8 waves; wave w owns head w. Block handles one (batch, 16-query)
// tile and streams k in tiles of 32. Softmax normalizes across the 8 heads at
// each (q,k) element -> cross-wave exchange through double-buffered LDS
// (one barrier per k-iteration). Next k-tile is prefetched
// (global_prefetch_b8) while the current tile computes.

__global__ __launch_bounds__(256) void attn_kernel(
    const bf16_t* __restrict__ Qp, const bf16_t* __restrict__ Kp,
    const bf16_t* __restrict__ VpT, bf16_t* __restrict__ Ctx) {
  __shared__ float  lds_sc[2][N_HEAD * 512];  // double-buffered score exchange
  __shared__ bf16_t lds_at[N_HEAD * 512];     // per-wave attn staging (16x32)

  const int wave = threadIdx.x >> 5;          // == head index
  const int lane = threadIdx.x & 31;
  const int h    = wave;
  const int b    = blockIdx.x >> 7;           // 128 q-tiles per batch
  const int q0   = (blockIdx.x & 127) << 4;

  const bf16_t* Qb = Qp  + (size_t)b * S_LEN * D_MODEL + (size_t)h * D_HEAD;
  const bf16_t* Kb = Kp  + (size_t)b * S_LEN * D_MODEL + (size_t)h * D_HEAD;
  const bf16_t* Vt = VpT + (size_t)(b * N_HEAD + h) * D_HEAD * S_LEN;

  // Q fragments for this head, 16 queries x 128 dh (4 K-chunks), resident.
  v16bf qf[4];
  for (int c = 0; c < 4; ++c) qf[c] = frag_bf16(Qb, D_MODEL, q0, c * 32);

  v8f acc[8] = {};                            // ctx tile 16x128 f32
  const float rs = 0.08838834764831845f;      // 1/sqrt(128)

  for (int k0 = 0; k0 < S_LEN; k0 += 32) {
    const int pb = (k0 >> 5) & 1;

    // prefetch next k-tile: 32 lanes cover 32 rows of K (one row each) and
    // 32 of the 128 transposed-V rows (cacheline-granular, covers the tile)
    if (k0 + 32 < S_LEN) {
      __builtin_prefetch(Kb + (size_t)(k0 + 32 + lane) * D_MODEL, 0, 3);
      __builtin_prefetch(Vt + (size_t)(lane * 4) * S_LEN + (k0 + 32), 0, 3);
    }

    // scores: q_h (16x128) @ k_h^T -> two 16x16 tiles (k-positions)
    v8f sc[2];
    for (int nt = 0; nt < 2; ++nt) {
      v8f s = {};
      for (int c = 0; c < 4; ++c) {
        v16bf kf = frag_bf16(Kb, D_MODEL, k0 + nt * 16, c * 32);
        s = __builtin_amdgcn_wmma_f32_16x16x32_bf16(
            false, qf[c], false, kf, (short)0, s, false, false);
      }
      sc[nt] = s;
    }
    // publish scaled scores for cross-head reduction
    for (int nt = 0; nt < 2; ++nt)
      for (int r = 0; r < 8; ++r)
        lds_sc[pb][wave * 512 + nt * 256 + r * 32 + lane] = sc[nt][r] * rs;
    __syncthreads();  // also separates prior-iter reads of this buffer

    // online max / sum(exp) across the 8 heads at this (q,k) element
    for (int nt = 0; nt < 2; ++nt) {
      for (int r = 0; r < 8; ++r) {
        float m = -1e30f, ssum = 0.f;
        for (int hh = 0; hh < N_HEAD; ++hh) {
          float v = lds_sc[pb][hh * 512 + nt * 256 + r * 32 + lane];
          if (v > m) { ssum = ssum * __expf(m - v) + 1.f; m = v; }
          else       { ssum += __expf(v - m); }
        }
        float a = __expf(sc[nt][r] * rs - m) / ssum;
        // D-layout element -> row-major staging (wave-private, same-wave DS order)
        const int mr  = r + ((lane >> 4) << 3);
        const int col = nt * 16 + (lane & 15);
        lds_at[wave * 512 + mr * 32 + col] = (bf16_t)a;
      }
    }

    // reload attn as a bf16 A-fragment (16x32); wave-private, no barrier
    v16bf af;
    {
      const int r  = lane & 15;
      const int kb = (lane >> 4) << 3;
      const bf16_t* p = &lds_at[wave * 512 + r * 32 + kb];
      V16BF u;
      u.h[0] = *(const v8bf*)(p);
      u.h[1] = *(const v8bf*)(p + 16);
      af = u.v;
    }

    // ctx += attn (16x32) @ v_h (32x128); V is head-transposed [dh][s], so
    // each B-fragment is the standard contiguous two-16B-load pattern.
    for (int t = 0; t < 8; ++t) {
      v16bf vf = frag_bf16(Vt, S_LEN, t * 16, k0);
      acc[t] = __builtin_amdgcn_wmma_f32_16x16x32_bf16(
          false, af, false, vf, (short)0, acc[t], false, false);
    }
  }

  // store ctx tile (heads concatenated along the feature dim)
  bf16_t* Cb = Ctx + (size_t)b * S_LEN * D_MODEL + (size_t)h * D_HEAD;
  for (int t = 0; t < 8; ++t)
    for (int r = 0; r < 8; ++r) {
      const int mr  = q0 + r + ((lane >> 4) << 3);
      const int col = t * 16 + (lane & 15);
      Cb[(size_t)mr * D_MODEL + col] = (bf16_t)acc[t][r];
    }
}

// ---------------------------------------------------------------------------

extern "C" void kernel_launch(void* const* d_in, const int* in_sizes, int n_in,
                              void* d_out, int out_size, void* d_ws, size_t ws_size,
                              hipStream_t stream) {
  (void)in_sizes; (void)n_in; (void)out_size; (void)ws_size;
  const float* Q  = (const float*)d_in[0];
  const float* K  = (const float*)d_in[1];
  const float* V  = (const float*)d_in[2];
  const float* WQ = (const float*)d_in[3];
  const float* WK = (const float*)d_in[4];
  const float* WV = (const float*)d_in[5];
  const float* WO = (const float*)d_in[6];

  // workspace: Qp, Kp, VpT, Ctx in bf16 (16 MiB each, 64 MiB total)
  const size_t mat = (size_t)BS * D_MODEL;
  bf16_t* Qp  = (bf16_t*)d_ws;
  bf16_t* Kp  = Qp + mat;
  bf16_t* VpT = Kp + mat;
  bf16_t* Ctx = VpT + mat;

  dim3 blk(256);
  dim3 gG(D_MODEL / 64, BS / 128);                 // (16, 64)
  gemm_nt_f32_to_bf16   <<<gG, blk, 0, stream>>>(Q, WQ, Qp);
  gemm_nt_f32_to_bf16   <<<gG, blk, 0, stream>>>(K, WK, Kp);
  gemm_nt_f32_to_bf16_vT<<<gG, blk, 0, stream>>>(V, WV, VpT);

  attn_kernel<<<dim3(N_BATCH * S_LEN / 16), blk, 0, stream>>>(Qp, Kp, VpT, Ctx);

  gemm_nt_bf16_to_f32<<<gG, blk, 0, stream>>>(Ctx, WO, (float*)d_out);
}
